// Generator_73400991089363
// MI455X (gfx1250) — matrix operand order repaired and checked by reading, not compiled, not run on hardware
//
#include <hip/hip_runtime.h>
#include <stdint.h>
#include <stddef.h>

// Problem constants (from reference)
#define HH 450
#define HP 480           // H padded to multiple of 32 for WMMA K/N tiling
#define MM 50000         // N_MESS
#define KNEI 6
#define NROOTS 1000
#define DEPTH_ 5
#define KT (HP / 32)     // 15 K-tiles per GEMM
#define NTILES (HP / 16) // 30
#define MTILES (MM / 16) // 3125

typedef __attribute__((ext_vector_type(16))) _Float16 v16h;
typedef __attribute__((ext_vector_type(8)))  float    v8f;

__device__ __forceinline__ float sigmf(float t) { return 1.0f / (1.0f + __expf(-t)); }

// ---------------------------------------------------------------------------
// Weight transpose + pad to f16: Bt[k][n] = W[n][k], zeros outside [HH,HH].
// blockIdx.y selects matrix: 0=Wz_a 1=Wz_b 2=Wr 3=Ur 4=Wh_a 5=Wh_b
// ---------------------------------------------------------------------------
__global__ __launch_bounds__(256)
void wtrans_kernel(const float* __restrict__ Wz, const float* __restrict__ Wr,
                   const float* __restrict__ Ur, const float* __restrict__ Wh,
                   _Float16* __restrict__ wts)
{
    int idx = blockIdx.x * 256 + threadIdx.x;
    if (idx >= HP * HP) return;
    int i = idx / HP;   // input dim (K)
    int o = idx % HP;   // output dim (N)
    int mid = blockIdx.y;
    float v = 0.0f;
    if (i < HH && o < HH) {
        switch (mid) {
            case 0: v = Wz[(size_t)o * (2 * HH) + i]; break;
            case 1: v = Wz[(size_t)o * (2 * HH) + HH + i]; break;
            case 2: v = Wr[(size_t)o * HH + i]; break;
            case 3: v = Ur[(size_t)o * HH + i]; break;
            case 4: v = Wh[(size_t)o * (2 * HH) + i]; break;
            case 5: v = Wh[(size_t)o * (2 * HH) + HH + i]; break;
        }
    }
    wts[(size_t)mid * HP * HP + (size_t)i * HP + o] = (_Float16)v;
}

// ---------------------------------------------------------------------------
// x[m,:] = emb[fnode[fmess[m]], :] as f16, padded with zeros to HP
// ---------------------------------------------------------------------------
__global__ __launch_bounds__(256)
void xbuild_kernel(const int* __restrict__ fnode, const int* __restrict__ fmess,
                   const float* __restrict__ emb, _Float16* __restrict__ xf)
{
    int m = blockIdx.x;
    int d = threadIdx.x;
    if (d >= HP / 2) return;
    int o = 2 * d;
    size_t base = (size_t)m * HP + o;
    if (o >= HH) { xf[base] = (_Float16)0.0f; xf[base + 1] = (_Float16)0.0f; return; }
    int v = fnode[fmess[m]];
    const float* e = emb + (size_t)v * HH + o;
    xf[base]     = (_Float16)e[0];
    xf[base + 1] = (_Float16)e[1];
}

__global__ __launch_bounds__(256)
void zero_kernel(uint32_t* __restrict__ p, long n)
{
    long i = (long)blockIdx.x * 256 + threadIdx.x;
    if (i < n) p[i] = 0u;
}

// ---------------------------------------------------------------------------
// Fragment load helpers (CDNA5 ISA 7.12.2 wave32 layouts).
// A 16x32 f16: lanes 0-15 hold K 0-7 & 16-23; lanes 16-31 hold K 8-15 & 24-31.
// B 32x16 f16: lane holds column n = lane%16, K in the same order as A.
// ---------------------------------------------------------------------------
union fragu { v16h v; uint32_t u[8]; };

__device__ __forceinline__ v16h load_afrag(const _Float16* Arow, int k0, int hi)
{
    fragu af;
    const uint32_t* ap = (const uint32_t*)(Arow + k0 + 8 * hi);
#pragma unroll
    for (int e = 0; e < 4; ++e) af.u[e] = ap[e];
#pragma unroll
    for (int e = 0; e < 4; ++e) af.u[4 + e] = ap[8 + e];
    return af.v;
}

__device__ __forceinline__ v16h load_bfrag(const _Float16* Bt, int ntile, int k0, int lane)
{
    v16h bf;
    const _Float16* bp = Bt + ntile * 16 + (lane & 15) + (size_t)(k0 + 8 * (lane >> 4)) * HP;
#pragma unroll
    for (int e = 0; e < 8; ++e) { bf[e] = bp[(size_t)e * HP]; bf[8 + e] = bp[(size_t)(16 + e) * HP]; }
    return bf;
}

// ---------------------------------------------------------------------------
// Generic WMMA GEMM, register-blocked over NPW n-tiles per wave.
// out[m,n] = sum_k A[m,k]*Bt[k,n] (+ bias[n]); f32 and/or f16 output.
// Waves in a block share the n-group and take consecutive m-tiles (B reuse in WGP$).
// ---------------------------------------------------------------------------
template <int NPW>
__global__ __launch_bounds__(256)
void gemm_kernel(const _Float16* __restrict__ A, const _Float16* __restrict__ Bt,
                 const float* __restrict__ bias,
                 float* __restrict__ outF32, _Float16* __restrict__ outF16)
{
    constexpr int NG = NTILES / NPW;
    int wave = (blockIdx.x * 256 + threadIdx.x) >> 5;
    if (wave >= MTILES * NG) return;            // wave-uniform: EXEC all-1s inside
    int lane = threadIdx.x & 31;
    int r  = lane & 15;
    int hi = lane >> 4;
    int mt = wave % MTILES;                      // consecutive m per block
    int ng = wave / MTILES;

    const _Float16* Arow = A + (size_t)(mt * 16 + r) * HP;

    v8f acc[NPW];
#pragma unroll
    for (int j = 0; j < NPW; ++j) acc[j] = (v8f){};

    for (int kt = 0; kt < KT; ++kt) {
        int k0 = kt * 32;
        v16h af = load_afrag(Arow, k0, hi);
#pragma unroll
        for (int j = 0; j < NPW; ++j) {
            v16h bf = load_bfrag(Bt, ng * NPW + j, k0, lane);
            acc[j] = __builtin_amdgcn_wmma_f32_16x16x32_f16(
                false, af, false, bf, (short)0, acc[j], false, false);
        }
    }

    // C/D layout: n = lane&15 ; m = i + 8*(lane>=16)
#pragma unroll
    for (int j = 0; j < NPW; ++j) {
        int n = (ng * NPW + j) * 16 + r;
        float bval = (bias && n < HH) ? bias[n] : 0.0f;
#pragma unroll
        for (int i = 0; i < 8; ++i) {
            int m = mt * 16 + i + 8 * hi;
            size_t off = (size_t)m * HP + n;
            float v = acc[j][i] + bval;
            if (outF32) outF32[off] = v;
            if (outF16) outF16[off] = (_Float16)v;
        }
    }
}

// ---------------------------------------------------------------------------
// Fused z/h GEMMs + GRU update:
//   accz = sum_h @ Wz_b.T ; acch = sum_gated @ Wh_b.T
//   z = sigmoid(accz + xz) ; pre_h = tanh(acch + xh)
//   h = (1-z)*sum_h + z*pre_h  (row 0 and padded columns forced to 0)
// ---------------------------------------------------------------------------
__global__ __launch_bounds__(256)
void zh_fused_kernel(const _Float16* __restrict__ Asumh, const _Float16* __restrict__ Asumg,
                     const _Float16* __restrict__ BtWzb, const _Float16* __restrict__ BtWhb,
                     const float* __restrict__ xz, const float* __restrict__ xh,
                     _Float16* __restrict__ hf)
{
    constexpr int NPW = 3;
    constexpr int NG = NTILES / NPW;            // 10
    int wave = (blockIdx.x * 256 + threadIdx.x) >> 5;
    if (wave >= MTILES * NG) return;
    int lane = threadIdx.x & 31;
    int r  = lane & 15;
    int hi = lane >> 4;
    int mt = wave % MTILES;
    int ng = wave / MTILES;

    const _Float16* Azrow = Asumh + (size_t)(mt * 16 + r) * HP;
    const _Float16* Agrow = Asumg + (size_t)(mt * 16 + r) * HP;

    v8f accz[NPW], acch[NPW];
#pragma unroll
    for (int j = 0; j < NPW; ++j) { accz[j] = (v8f){}; acch[j] = (v8f){}; }

    for (int kt = 0; kt < KT; ++kt) {
        int k0 = kt * 32;
        v16h afz = load_afrag(Azrow, k0, hi);
        v16h afg = load_afrag(Agrow, k0, hi);
#pragma unroll
        for (int j = 0; j < NPW; ++j) {
            v16h bz = load_bfrag(BtWzb, ng * NPW + j, k0, lane);
            accz[j] = __builtin_amdgcn_wmma_f32_16x16x32_f16(
                false, afz, false, bz, (short)0, accz[j], false, false);
            v16h bh = load_bfrag(BtWhb, ng * NPW + j, k0, lane);
            acch[j] = __builtin_amdgcn_wmma_f32_16x16x32_f16(
                false, afg, false, bh, (short)0, acch[j], false, false);
        }
    }

#pragma unroll
    for (int j = 0; j < NPW; ++j) {
        int n = (ng * NPW + j) * 16 + r;
#pragma unroll
        for (int i = 0; i < 8; ++i) {
            int m = mt * 16 + i + 8 * hi;
            size_t off = (size_t)m * HP + n;
            float hv = 0.0f;
            if (n < HH && m != 0) {
                float z  = sigmf(accz[j][i] + xz[off]);
                float ph = tanhf(acch[j][i] + xh[off]);
                float s  = (float)Asumh[off];
                hv = (1.0f - z) * s + z * ph;
            }
            hf[off] = (_Float16)hv;
        }
    }
}

// ---------------------------------------------------------------------------
// sum_h[m,:] = sum_k h[mess_graph[m,k], :]   (fp32 accumulate, f16 store)
// ---------------------------------------------------------------------------
__global__ __launch_bounds__(256)
void sumh_kernel(const _Float16* __restrict__ hf, const int* __restrict__ mg,
                 _Float16* __restrict__ sumh)
{
    int m = blockIdx.x;
    int d = threadIdx.x;
    if (d >= HP / 2) return;
    int o = 2 * d;
    float a0 = 0.0f, a1 = 0.0f;
#pragma unroll
    for (int k = 0; k < KNEI; ++k) {
        int j = mg[m * KNEI + k];
        const _Float16* p = hf + (size_t)j * HP + o;
        a0 += (float)p[0];
        a1 += (float)p[1];
    }
    _Float16* q = sumh + (size_t)m * HP + o;
    q[0] = (_Float16)a0;
    q[1] = (_Float16)a1;
}

// ---------------------------------------------------------------------------
// sum_gated[m,:] = sum_k sigmoid(xr[m,:] + hUr[j,:]) * h[j,:],  j=mess_graph[m,k]
// ---------------------------------------------------------------------------
__global__ __launch_bounds__(256)
void sumg_kernel(const _Float16* __restrict__ hf, const _Float16* __restrict__ hur,
                 const _Float16* __restrict__ xr, const int* __restrict__ mg,
                 _Float16* __restrict__ sumg)
{
    int m = blockIdx.x;
    int d = threadIdx.x;
    if (d >= HP / 2) return;
    int o = 2 * d;
    size_t mo = (size_t)m * HP + o;
    float x0 = (float)xr[mo], x1 = (float)xr[mo + 1];
    float a0 = 0.0f, a1 = 0.0f;
#pragma unroll
    for (int k = 0; k < KNEI; ++k) {
        int j = mg[m * KNEI + k];
        size_t jo = (size_t)j * HP + o;
        float u0 = (float)hur[jo], u1 = (float)hur[jo + 1];
        float h0 = (float)hf[jo],  h1 = (float)hf[jo + 1];
        a0 += sigmf(x0 + u0) * h0;
        a1 += sigmf(x1 + u1) * h1;
    }
    sumg[mo]     = (_Float16)a0;
    sumg[mo + 1] = (_Float16)a1;
}

// ---------------------------------------------------------------------------
// out[i, 0:450] = emb[fnode[root]] ; out[i, 450:900] = sum_k h[node_graph[root,k]]
// ---------------------------------------------------------------------------
__global__ __launch_bounds__(256)
void out_kernel(const int* __restrict__ fnode, const int* __restrict__ root,
                const int* __restrict__ ng, const float* __restrict__ emb,
                const _Float16* __restrict__ hf, float* __restrict__ out)
{
    int i = blockIdx.x;
    int rt = root[i];
    for (int j = threadIdx.x; j < 2 * HH; j += blockDim.x) {
        float v;
        if (j < HH) {
            v = emb[(size_t)fnode[rt] * HH + j];
        } else {
            int o = j - HH;
            float a = 0.0f;
#pragma unroll
            for (int k = 0; k < KNEI; ++k) {
                int msg = ng[rt * KNEI + k];
                a += (float)hf[(size_t)msg * HP + o];
            }
            v = a;
        }
        out[(size_t)i * (2 * HH) + j] = v;
    }
}

// ---------------------------------------------------------------------------
extern "C" void kernel_launch(void* const* d_in, const int* in_sizes, int n_in,
                              void* d_out, int out_size, void* d_ws, size_t ws_size,
                              hipStream_t stream)
{
    const int*   fnode = (const int*)d_in[0];
    const int*   fmess = (const int*)d_in[1];
    const int*   ngrf  = (const int*)d_in[2];   // node_graph
    const int*   mg    = (const int*)d_in[3];   // mess_graph
    const int*   root  = (const int*)d_in[4];
    // d_in[5] = depth (device scalar); fixed at DEPTH_=5 per problem definition
    const float* emb = (const float*)d_in[6];
    const float* Wz  = (const float*)d_in[7];
    const float* bz  = (const float*)d_in[8];
    const float* Wr  = (const float*)d_in[9];
    const float* br  = (const float*)d_in[10];
    const float* Ur  = (const float*)d_in[11];
    const float* Wh  = (const float*)d_in[12];
    const float* bh  = (const float*)d_in[13];
    float* out = (float*)d_out;

    // Workspace carve-up (256B aligned)
    char* ws = (char*)d_ws;
    size_t off = 0;
    auto alloc = [&](size_t bytes) -> void* {
        void* p = ws + off;
        off += (bytes + 255) & ~(size_t)255;
        return p;
    };
    const size_t F16MAT = (size_t)MM * HP * sizeof(_Float16);
    const size_t F32MAT = (size_t)MM * HP * sizeof(float);
    _Float16* xf   = (_Float16*)alloc(F16MAT);
    _Float16* hf   = (_Float16*)alloc(F16MAT);
    _Float16* hur  = (_Float16*)alloc(F16MAT);
    _Float16* xrf  = (_Float16*)alloc(F16MAT);
    _Float16* sumh = (_Float16*)alloc(F16MAT);
    _Float16* sumg = (_Float16*)alloc(F16MAT);
    float*    xz   = (float*)alloc(F32MAT);
    float*    xh   = (float*)alloc(F32MAT);
    _Float16* wts  = (_Float16*)alloc((size_t)6 * HP * HP * sizeof(_Float16));
    (void)ws_size; (void)in_sizes; (void)n_in; (void)out_size;

    auto Bmat = [&](int mid) { return wts + (size_t)mid * HP * HP; };

    constexpr int NPW_G = 5;                               // generic GEMM n-blocking
    const int GEMM_BLOCKS  = (MTILES * (NTILES / NPW_G) + 7) / 8;   // 8 waves / block
    const int FUSED_BLOCKS = (MTILES * (NTILES / 3) + 7) / 8;

    // --- setup ---
    wtrans_kernel<<<dim3((HP * HP + 255) / 256, 6), 256, 0, stream>>>(Wz, Wr, Ur, Wh, wts);
    xbuild_kernel<<<MM, 256, 0, stream>>>(fnode, fmess, emb, xf);
    {
        long ndw = (long)MM * HP / 2;
        zero_kernel<<<(unsigned)((ndw + 255) / 256), 256, 0, stream>>>((uint32_t*)hf, ndw);
    }

    // --- loop-invariant precomputes: xz = x@Wz_a.T+bz, xr = x@Wr.T+br, xh = x@Wh_a.T+bh
    gemm_kernel<NPW_G><<<GEMM_BLOCKS, 256, 0, stream>>>(xf, Bmat(0), bz, xz, nullptr);
    gemm_kernel<NPW_G><<<GEMM_BLOCKS, 256, 0, stream>>>(xf, Bmat(2), br, nullptr, xrf);
    gemm_kernel<NPW_G><<<GEMM_BLOCKS, 256, 0, stream>>>(xf, Bmat(4), bh, xh, nullptr);

    // --- depth-unrolled GRU message passing ---
    for (int it = 0; it < DEPTH_; ++it) {
        sumh_kernel<<<MM, 256, 0, stream>>>(hf, mg, sumh);
        // hUr = h @ Ur.T
        gemm_kernel<NPW_G><<<GEMM_BLOCKS, 256, 0, stream>>>(hf, Bmat(3), nullptr, nullptr, hur);
        sumg_kernel<<<MM, 256, 0, stream>>>(hf, hur, xrf, mg, sumg);
        // fused: z/h GEMMs + gate nonlinearities + GRU update + mask -> new h (f16)
        zh_fused_kernel<<<FUSED_BLOCKS, 256, 0, stream>>>(sumh, sumg, Bmat(1), Bmat(5),
                                                          xz, xh, hf);
    }

    // --- root readout ---
    out_kernel<<<NROOTS, 256, 0, stream>>>(fnode, root, ngrf, emb, hf, out);
}
